// MACE_layer_60851096650036
// MI455X (gfx1250) — matrix operand order repaired and checked by reading, not compiled, not run on hardware
//
#include <hip/hip_runtime.h>

// ---------------------------------------------------------------------------
// MACE-style equivariant GNN layer for MI455X (gfx1250), wave32 + bf16 WMMA.
// All GEMMs use v_wmma_f32_16x16x32_bf16 with fp32 accumulation.
// Edge-MLP weights are staged to LDS with global_load_async_to_lds_b128.
// ---------------------------------------------------------------------------

typedef __bf16 bf16;
typedef __attribute__((ext_vector_type(16))) __bf16 v16bf;
typedef __attribute__((ext_vector_type(8)))  __bf16 v8bf;
typedef __attribute__((ext_vector_type(8)))  float   v8f;

#define WAIT_DS()    asm volatile("s_wait_dscnt 0" ::: "memory")
#define WAIT_ASYNC() asm volatile("s_wait_asynccnt 0" ::: "memory")

// bf16 weight pool offsets (in bf16 elements); all matrices stored
// column-major with K=64 rows so B-fragment loads are contiguous.
// NOTE: W1 and W2 are contiguous so the edge kernel can async-stage both
// with one 48KB range.
enum : int {
  OFF_WSUP   = 0,
  OFF_WVUP   = 4096,
  OFF_W1     = 8192,
  OFF_W2     = 12288,   // 64 x 320
  OFF_WSPOST = 32768,
  OFF_WVPOST = 36864,
  OFF_PS1    = 40960,
  OFF_PS2    = 45056,
  OFF_PS3    = 49152,
  OFF_PVV    = 53248,
  OFF_PV1    = 57344,
  OFF_PV2    = 61440,
  OFF_PV3    = 65536,
  OFF_RW1    = 69632,
  OFF_RW2    = 73728,
  BF_TOTAL   = 77824
};
#define BF_BYTES (BF_TOTAL * 2)

__device__ __forceinline__ float silu_f(float x) { return x / (1.0f + __expf(-x)); }

__device__ __forceinline__ v8f wmma_bf16(v16bf a, v16bf b, v8f c) {
  return __builtin_amdgcn_wmma_f32_16x16x32_bf16(false, a, false, b, (short)0, c,
                                                 false, false);
}

// A fragment (16x32) from fp32 row-major source. Lane (m, m+16) holds row m.
// Slot j<8 -> K = kb + hi*8 + j ; slot j>=8 -> K = kb + 16 + hi*8 + (j-8).
__device__ __forceinline__ v16bf afrag_f32(const float* row, int kb, int hi) {
  v8f x0 = *(const v8f*)(row + kb + hi * 8);
  v8f x1 = *(const v8f*)(row + kb + 16 + hi * 8);
  v16bf a;
#pragma unroll
  for (int j = 0; j < 8; ++j) { a[j] = (bf16)x0[j]; a[j + 8] = (bf16)x1[j]; }
  return a;
}

// A fragment with strided fp32 K (for the (C,3)-interleaved vector features).
__device__ __forceinline__ v16bf afrag_strided_f32(const float* base, int stride,
                                                   int kb, int hi) {
  v16bf a;
  int c0 = kb + hi * 8;
#pragma unroll
  for (int j = 0; j < 8; ++j) {
    a[j]     = (bf16)base[(c0 + j) * stride];
    a[j + 8] = (bf16)base[(c0 + 16 + j) * stride];
  }
  return a;
}

// A fragment from bf16 row-major LDS slab (row length 64).
__device__ __forceinline__ v16bf afrag_lds_bf16(const bf16* row, int kb, int hi) {
  v8bf x0 = *(const v8bf*)(row + kb + hi * 8);
  v8bf x1 = *(const v8bf*)(row + kb + 16 + hi * 8);
  v16bf a;
#pragma unroll
  for (int j = 0; j < 8; ++j) { a[j] = x0[j]; a[j + 8] = x1[j]; }
  return a;
}

// B fragment (32x16) from bf16 column-major weights (64 rows per column).
// Lane (n, n+16) holds column n; slot j<8 -> K = kb + hi*8 + j ; slot j>=8 -> K = kb + 16 + hi*8 + (j-8).
__device__ __forceinline__ v16bf bfrag(const bf16* Wcm, int col, int kb, int hi) {
  const bf16* p = Wcm + (size_t)col * 64 + kb + hi * 8;
  v8bf x0 = *(const v8bf*)(p);
  v8bf x1 = *(const v8bf*)(p + 16);
  v16bf b;
#pragma unroll
  for (int j = 0; j < 8; ++j) { b[j] = x0[j]; b[j + 8] = x1[j]; }
  return b;
}

// D tile -> row-major fp32 store (global or LDS). VGPR r is (M=r+hi*8, N=lane%16).
__device__ __forceinline__ void store_tile(float* dst, int ld, const v8f& d,
                                           int colbase, int lr, int hi) {
#pragma unroll
  for (int r = 0; r < 8; ++r) dst[(size_t)(r + hi * 8) * ld + colbase + lr] = d[r];
}

// ---------------------------------------------------------------------------
// Weight conversion: fp32 row-major (64 x cols) -> bf16 column-major.
// ---------------------------------------------------------------------------
__global__ void convert_wcm(const float* __restrict__ src, bf16* __restrict__ dst,
                            int cols) {
  int i = blockIdx.x * blockDim.x + threadIdx.x;
  if (i >= cols * 64) return;
  int col = i >> 6, k = i & 63;
  dst[i] = (bf16)src[k * cols + col];
}

// ---------------------------------------------------------------------------
// Node up-projection: s_up = nf[:, :64] @ W_s_up ;
// v_up[i] = nf[:, 64 + 3c + i] @ W_v_up  (3 planes, each N x 64 row-major).
// One wave per 16-node tile, 4 waves per block.
// ---------------------------------------------------------------------------
__global__ __launch_bounds__(128) void up_kernel(const float* __restrict__ nf,
                                                 const bf16* __restrict__ wbf,
                                                 float* __restrict__ s_up,
                                                 float* __restrict__ v_up, int Nn) {
  int wave = threadIdx.x >> 5;
  int tile = blockIdx.x * 4 + wave;
  if (tile * 16 >= Nn) return;
  int lane = threadIdx.x & 31, lr = lane & 15, hi = lane >> 4;
  int n = tile * 16 + lr;
  const v8f vz = {0, 0, 0, 0, 0, 0, 0, 0};

  {  // scalar channel
    v8f acc[4] = {vz, vz, vz, vz};
#pragma unroll
    for (int ks = 0; ks < 2; ++ks) {
      int kb = ks * 32;
      v16bf a = afrag_f32(nf + (size_t)n * 256, kb, hi);
#pragma unroll
      for (int t = 0; t < 4; ++t)
        acc[t] = wmma_bf16(a, bfrag(wbf + OFF_WSUP, t * 16 + lr, kb, hi), acc[t]);
    }
#pragma unroll
    for (int t = 0; t < 4; ++t)
      store_tile(s_up + (size_t)tile * 16 * 64, 64, acc[t], t * 16, lr, hi);
  }
#pragma unroll
  for (int i = 0; i < 3; ++i) {  // vector planes
    v8f acc[4] = {vz, vz, vz, vz};
#pragma unroll
    for (int ks = 0; ks < 2; ++ks) {
      int kb = ks * 32;
      v16bf a = afrag_strided_f32(nf + (size_t)n * 256 + 64 + i, 3, kb, hi);
#pragma unroll
      for (int t = 0; t < 4; ++t)
        acc[t] = wmma_bf16(a, bfrag(wbf + OFF_WVUP, t * 16 + lr, kb, hi), acc[t]);
    }
#pragma unroll
    for (int t = 0; t < 4; ++t)
      store_tile(v_up + (size_t)i * Nn * 64 + (size_t)tile * 16 * 64, 64, acc[t],
                 t * 16, lr, hi);
  }
}

// ---------------------------------------------------------------------------
// Edge kernel (persistent over ET tiles per wave, 2 waves per block):
//   stage W1|W2 (48KB bf16) to LDS via global_load_async_to_lds_b128, then per
//   16-edge tile:
//     h = silu(ef @ W1)   (8 WMMA, B from LDS, D->LDS bf16 transpose)
//     w = h @ W2          (40 WMMA, B from LDS, 20 live D accumulators)
//     messages consumed DIRECTLY from the D registers (w0..w4 for a given
//     (edge,channel) are the same lane/slot across 5 accumulator tiles),
//     then atomic scatter a_s += m_s/16, a_v += m_v/16.
// ---------------------------------------------------------------------------
#define EW 2   // waves per block
#define ET 8   // edge tiles per wave
__global__ __launch_bounds__(64) void edge_kernel(
    const float* __restrict__ vectors, const float* __restrict__ ef,
    const int* __restrict__ eidx, const float* __restrict__ s_up,
    const float* __restrict__ v_up, const bf16* __restrict__ wbf,
    float* __restrict__ a_s, float* __restrict__ a_v, int Ne, int Nn) {
  // [0, 4096) = W1 (64x64 cm), [4096, 24576) = W2 (64x320 cm)
  __shared__ __align__(64) bf16 wsh[4096 + 20480];   // 48 KB
  __shared__ __align__(32) bf16 h_lds[EW][16 * 64];  //  4 KB

  int wave = threadIdx.x >> 5;
  int lane = threadIdx.x & 31, lr = lane & 15, hi = lane >> 4;
  int tid = threadIdx.x;

  // ---- async stage W1|W2 -> LDS (each instruction: 32 lanes x 16B) ----
  {
    uint64_t g = (uint64_t)(uintptr_t)(wbf + OFF_W1);
    uint32_t l = (uint32_t)(uintptr_t)(void*)wsh;
#pragma unroll 4
    for (int it = 0; it < 48; ++it) {  // 48 x 1KB (both waves together)
      uint32_t loff = l + (uint32_t)(tid * 16 + it * 1024);
      uint64_t goff = g + (uint64_t)(tid * 16 + it * 1024);
      asm volatile("global_load_async_to_lds_b128 %0, %1, off"
                   :: "v"(loff), "v"(goff) : "memory");
    }
    WAIT_ASYNC();
    __syncthreads();
  }
  const bf16* w1s = wsh;
  const bf16* w2s = wsh + 4096;
  const v8f vz = {0, 0, 0, 0, 0, 0, 0, 0};

  int tileBase = (blockIdx.x * EW + wave) * ET;
  for (int it = 0; it < ET; ++it) {
    int eBase = (tileBase + it) * 16;
    if (eBase >= Ne) break;  // wave-uniform
    int e = eBase + lr;
    if (eBase + 16 < Ne) __builtin_prefetch(ef + (size_t)(e + 16) * 64, 0);

    // ---- h = silu(ef @ W1) ----
    v8f hacc[4] = {vz, vz, vz, vz};
#pragma unroll
    for (int ks = 0; ks < 2; ++ks) {
      int kb = ks * 32;
      v16bf a = afrag_f32(ef + (size_t)e * 64, kb, hi);
#pragma unroll
      for (int t = 0; t < 4; ++t)
        hacc[t] = wmma_bf16(a, bfrag(w1s, t * 16 + lr, kb, hi), hacc[t]);
    }
#pragma unroll
    for (int t = 0; t < 4; ++t)
#pragma unroll
      for (int r = 0; r < 8; ++r)
        h_lds[wave][(r + hi * 8) * 64 + t * 16 + lr] = (bf16)silu_f(hacc[t][r]);
    WAIT_DS();

    // ---- w = h @ W2 : 20 live accumulator tiles ----
    v16bf aw0 = afrag_lds_bf16(h_lds[wave] + lr * 64, 0, hi);
    v16bf aw1 = afrag_lds_bf16(h_lds[wave] + lr * 64, 32, hi);
    v8f wacc[20];
#pragma unroll
    for (int ct = 0; ct < 20; ++ct) {
      v8f acc = wmma_bf16(aw0, bfrag(w2s, ct * 16 + lr, 0, hi), vz);
      wacc[ct] = wmma_bf16(aw1, bfrag(w2s, ct * 16 + lr, 32, hi), acc);
    }

    // ---- messages straight from D registers ----
    // wacc[j*4+q], VGPR r, this lane  ==  w[e = eBase+r+hi*8][j*64 + q*16 + lr]
#pragma unroll
    for (int r = 0; r < 8; ++r) {
      int ee = eBase + r + hi * 8;
      int snd = eidx[ee], rcv = eidx[Ne + ee];
      float rx = vectors[(size_t)ee * 3], ry = vectors[(size_t)ee * 3 + 1],
            rz = vectors[(size_t)ee * 3 + 2];
      float inv = 1.7320508075688772f /
                  (__fsqrt_rn(rx * rx + ry * ry + rz * rz) + 1e-12f);
      float Y0 = rx * inv, Y1 = ry * inv, Y2 = rz * inv;
      const float* sjr  = s_up + (size_t)snd * 64;
      const float* vj0r = v_up + (size_t)snd * 64;
      const float* vj1r = vj0r + (size_t)Nn * 64;
      const float* vj2r = vj1r + (size_t)Nn * 64;
      float* as_r = a_s + (size_t)rcv * 64;
      float* av0r = a_v + (size_t)rcv * 64;
      float* av1r = av0r + (size_t)Nn * 64;
      float* av2r = av1r + (size_t)Nn * 64;
#pragma unroll
      for (int q = 0; q < 4; ++q) {
        int c = q * 16 + lr;
        float w0 = wacc[q][r];
        float w1 = wacc[4 + q][r];
        float w2 = wacc[8 + q][r];
        float w3 = wacc[12 + q][r];
        float w4 = wacc[16 + q][r];
        float sj = sjr[c];
        float u0 = vj0r[c], u1 = vj1r[c], u2 = vj2r[c];
        float vdY = u0 * Y0 + u1 * Y1 + u2 * Y2;
        float ms = (w0 * sj + w1 * vdY) * 0.0625f;
        float cx0 = u1 * Y2 - u2 * Y1;
        float cx1 = u2 * Y0 - u0 * Y2;
        float cx2 = u0 * Y1 - u1 * Y0;
        float ws2 = w2 * sj;
        float mv0 = (ws2 * Y0 + w3 * u0 + w4 * cx0) * 0.0625f;
        float mv1 = (ws2 * Y1 + w3 * u1 + w4 * cx1) * 0.0625f;
        float mv2 = (ws2 * Y2 + w3 * u2 + w4 * cx2) * 0.0625f;
        atomicAdd(as_r + c, ms);
        atomicAdd(av0r + c, mv0);
        atomicAdd(av1r + c, mv1);
        atomicAdd(av2r + c, mv2);
      }
    }
  }
}

// ---------------------------------------------------------------------------
// Fused node post block (one wave per 16-node tile):
//   as_post/av_post -> ps (P_s1..P_vv) -> pv (P_v1..P_v3)
//   hr = silu(ps @ R_w1), out_s = hr @ R_w2, gate/vec via VALU dots.
// ---------------------------------------------------------------------------
__global__ __launch_bounds__(32) void node_post_kernel(
    const float* __restrict__ a_s, const float* __restrict__ a_v,
    const bf16* __restrict__ wbf, const float* __restrict__ Rgate,
    const float* __restrict__ Rvmix, float* __restrict__ out, int Nn) {
  __shared__ __align__(32) float asb[16 * 64];
  __shared__ __align__(32) float avb[3][16 * 64];
  __shared__ __align__(32) float psb[16 * 64];
  __shared__ __align__(32) float pvb[3][16 * 64];
  __shared__ __align__(32) float hrb[16 * 64];
  __shared__ float gpart[32];

  int nBase = blockIdx.x * 16;
  int lane = threadIdx.x & 31, lr = lane & 15, hi = lane >> 4;
  const v8f vz = {0, 0, 0, 0, 0, 0, 0, 0};

  float* out_s = out;                                     // N x 64
  float* vec   = out + (size_t)Nn * 64;                   // N x 3
  float* nfo   = out + (size_t)Nn * 64 + (size_t)Nn * 3;  // N x 256

  // ---- Step 1: as_post = a_s @ W_s_post ; av_post_i = a_v_i @ W_v_post ----
  for (int p = 0; p < 4; ++p) {
    const float* src = (p == 0) ? a_s + (size_t)nBase * 64
                                : a_v + (size_t)(p - 1) * Nn * 64 + (size_t)nBase * 64;
    const bf16* W = (p == 0) ? wbf + OFF_WSPOST : wbf + OFF_WVPOST;
    float* dst = (p == 0) ? asb : avb[p - 1];
    v8f acc[4] = {vz, vz, vz, vz};
#pragma unroll
    for (int ks = 0; ks < 2; ++ks) {
      int kb = ks * 32;
      v16bf a = afrag_f32(src + lr * 64, kb, hi);
#pragma unroll
      for (int t = 0; t < 4; ++t)
        acc[t] = wmma_bf16(a, bfrag(W, t * 16 + lr, kb, hi), acc[t]);
    }
#pragma unroll
    for (int t = 0; t < 4; ++t) store_tile(dst, 64, acc[t], t * 16, lr, hi);
  }
  WAIT_DS();

  // ---- Step 2a: ps = a1@Ps1 + a2@Ps2 + a3@Ps3 + vv@Pvv ----
  v16bf fa1[2], fa2[2], fa3[2], fvv[2];
#pragma unroll
  for (int ks = 0; ks < 2; ++ks) {
    int c0 = ks * 32 + hi * 8;
    const float* ar = asb + lr * 64;
    const float* q0r = avb[0] + lr * 64;
    const float* q1r = avb[1] + lr * 64;
    const float* q2r = avb[2] + lr * 64;
#pragma unroll
    for (int j = 0; j < 16; ++j) {
      int c = c0 + (j < 8 ? j : 8 + j);  // slot->K mapping of A fragment
      float x = ar[c];
      fa1[ks][j] = (bf16)x;
      fa2[ks][j] = (bf16)(x * x);
      fa3[ks][j] = (bf16)(x * x * x);
      float q0 = q0r[c], q1 = q1r[c], q2 = q2r[c];
      fvv[ks][j] = (bf16)(q0 * q0 + q1 * q1 + q2 * q2);
    }
  }
  {
    v8f acc[4] = {vz, vz, vz, vz};
#pragma unroll
    for (int ks = 0; ks < 2; ++ks) {
      int kb = ks * 32;
#pragma unroll
      for (int t = 0; t < 4; ++t) {
        int col = t * 16 + lr;
        acc[t] = wmma_bf16(fa1[ks], bfrag(wbf + OFF_PS1, col, kb, hi), acc[t]);
        acc[t] = wmma_bf16(fa2[ks], bfrag(wbf + OFF_PS2, col, kb, hi), acc[t]);
        acc[t] = wmma_bf16(fa3[ks], bfrag(wbf + OFF_PS3, col, kb, hi), acc[t]);
        acc[t] = wmma_bf16(fvv[ks], bfrag(wbf + OFF_PVV, col, kb, hi), acc[t]);
      }
    }
#pragma unroll
    for (int t = 0; t < 4; ++t)
#pragma unroll
      for (int r = 0; r < 8; ++r) {
        int row = r + hi * 8, c = t * 16 + lr;
        float x = acc[t][r];
        psb[row * 64 + c] = x;
        nfo[(size_t)(nBase + row) * 256 + c] = x;
      }
  }

  // ---- Step 2b: pv_i = av@Pv1 + (as*av)@Pv2 + (as^2*av)@Pv3 ----
  for (int i = 0; i < 3; ++i) {
    v16bf fav[2], favs[2], favs2[2];
#pragma unroll
    for (int ks = 0; ks < 2; ++ks) {
      int c0 = ks * 32 + hi * 8;
      const float* ar = asb + lr * 64;
      const float* vr = avb[i] + lr * 64;
#pragma unroll
      for (int j = 0; j < 16; ++j) {
        int c = c0 + (j < 8 ? j : 8 + j);
        float s = ar[c], v = vr[c];
        fav[ks][j]   = (bf16)v;
        favs[ks][j]  = (bf16)(s * v);
        favs2[ks][j] = (bf16)(s * s * v);
      }
    }
    v8f acc[4] = {vz, vz, vz, vz};
#pragma unroll
    for (int ks = 0; ks < 2; ++ks) {
      int kb = ks * 32;
#pragma unroll
      for (int t = 0; t < 4; ++t) {
        int col = t * 16 + lr;
        acc[t] = wmma_bf16(fav[ks],   bfrag(wbf + OFF_PV1, col, kb, hi), acc[t]);
        acc[t] = wmma_bf16(favs[ks],  bfrag(wbf + OFF_PV2, col, kb, hi), acc[t]);
        acc[t] = wmma_bf16(favs2[ks], bfrag(wbf + OFF_PV3, col, kb, hi), acc[t]);
      }
    }
#pragma unroll
    for (int t = 0; t < 4; ++t)
#pragma unroll
      for (int r = 0; r < 8; ++r) {
        int row = r + hi * 8, c = t * 16 + lr;
        float x = acc[t][r];
        pvb[i][row * 64 + c] = x;
        nfo[(size_t)(nBase + row) * 256 + 64 + 3 * c + i] = x;
      }
  }
  WAIT_DS();

  // ---- Step 3: hr = silu(ps @ R_w1) ----
  {
    v8f acc[4] = {vz, vz, vz, vz};
#pragma unroll
    for (int ks = 0; ks < 2; ++ks) {
      int kb = ks * 32;
      v16bf a = afrag_f32(psb + lr * 64, kb, hi);
#pragma unroll
      for (int t = 0; t < 4; ++t)
        acc[t] = wmma_bf16(a, bfrag(wbf + OFF_RW1, t * 16 + lr, kb, hi), acc[t]);
    }
#pragma unroll
    for (int t = 0; t < 4; ++t)
#pragma unroll
      for (int r = 0; r < 8; ++r)
        hrb[(r + hi * 8) * 64 + t * 16 + lr] = silu_f(acc[t][r]);
  }
  WAIT_DS();

  // ---- out_s = hr @ R_w2 ----
  {
    v8f acc[4] = {vz, vz, vz, vz};
#pragma unroll
    for (int ks = 0; ks < 2; ++ks) {
      int kb = ks * 32;
      v16bf a = afrag_f32(hrb + lr * 64, kb, hi);
#pragma unroll
      for (int t = 0; t < 4; ++t)
        acc[t] = wmma_bf16(a, bfrag(wbf + OFF_RW2, t * 16 + lr, kb, hi), acc[t]);
    }
#pragma unroll
    for (int t = 0; t < 4; ++t)
      store_tile(out_s + (size_t)nBase * 64, 64, acc[t], t * 16, lr, hi);
  }

  // ---- gate = silu(hr @ R_gate) ; vec = (pv . R_vmix) * gate ----
  {
    float g = 0.f;
#pragma unroll 8
    for (int m = 0; m < 32; ++m) g += hrb[lr * 64 + hi * 32 + m] * Rgate[hi * 32 + m];
    gpart[lane] = g;
  }
  WAIT_DS();
#pragma unroll
  for (int it = 0; it < 2; ++it) {
    int idx = lane + it * 32;
    if (idx < 48) {
      int nn = idx / 3, i = idx - nn * 3;
      float g = silu_f(gpart[nn] + gpart[16 + nn]);
      float dot = 0.f;
#pragma unroll 8
      for (int c = 0; c < 64; ++c) dot += pvb[i][nn * 64 + c] * Rvmix[c];
      vec[(size_t)(nBase + nn) * 3 + i] = dot * g;
    }
  }
}

// ---------------------------------------------------------------------------
extern "C" void kernel_launch(void* const* d_in, const int* in_sizes, int n_in,
                              void* d_out, int out_size, void* d_ws, size_t ws_size,
                              hipStream_t stream) {
  (void)n_in; (void)out_size; (void)ws_size;
  const float* vectors = (const float*)d_in[0];
  // d_in[1] = lengths: unused by the reference.
  const float* nf   = (const float*)d_in[2];
  const float* ef   = (const float*)d_in[3];
  const int*   eidx = (const int*)d_in[4];
  const float* Rgate = (const float*)d_in[20];
  const float* Rvmix = (const float*)d_in[21];

  int Nn = in_sizes[2] / 256;  // 50000
  int Ne = in_sizes[3] / 64;   // 800000

  bf16*  wbf = (bf16*)d_ws;
  float* f32 = (float*)((char*)d_ws + BF_BYTES);
  float* s_up = f32;
  float* v_up = f32 + (size_t)Nn * 64;       // 3 planes
  float* a_s  = f32 + (size_t)4 * Nn * 64;
  float* a_v  = f32 + (size_t)5 * Nn * 64;   // 3 planes

  // Convert the 15 GEMM weight matrices to column-major bf16.
  static const int srcIdx[15] = {5, 6, 7, 8, 9, 10, 11, 12, 13, 14, 15, 16, 17, 18, 19};
  static const int off[15]    = {OFF_WSUP, OFF_WVUP, OFF_W1, OFF_W2, OFF_WSPOST,
                                 OFF_WVPOST, OFF_PS1, OFF_PS2, OFF_PS3, OFF_PVV,
                                 OFF_PV1, OFF_PV2, OFF_PV3, OFF_RW1, OFF_RW2};
  static const int cols[15]   = {64, 64, 64, 320, 64, 64, 64, 64, 64, 64, 64, 64, 64, 64, 64};
  for (int m = 0; m < 15; ++m) {
    int total = cols[m] * 64;
    convert_wcm<<<(total + 255) / 256, 256, 0, stream>>>(
        (const float*)d_in[srcIdx[m]], wbf + off[m], cols[m]);
  }

  // Zero the segment-sum accumulators (a_s followed contiguously by a_v).
  hipMemsetAsync(a_s, 0, (size_t)4 * Nn * 64 * sizeof(float), stream);

  int ntiles = (Nn + 15) / 16;
  up_kernel<<<(ntiles + 3) / 4, 128, 0, stream>>>(nf, wbf, s_up, v_up, Nn);

  int etiles = (Ne + 15) / 16;
  int ewaves = (etiles + ET - 1) / ET;
  int eblocks = (ewaves + EW - 1) / EW;
  edge_kernel<<<eblocks, 32 * EW, 0, stream>>>(vectors, ef, eidx, s_up, v_up, wbf,
                                               a_s, a_v, Ne, Nn);

  node_post_kernel<<<ntiles, 32, 0, stream>>>(a_s, a_v, wbf, Rgate, Rvmix,
                                              (float*)d_out, Nn);
}
